// LGAPEncoder_22935125360763
// MI455X (gfx1250) — matrix operand order repaired and checked by reading, not compiled, not run on hardware
//
#include <hip/hip_runtime.h>

typedef __attribute__((ext_vector_type(16))) _Float16 v16h;
typedef __attribute__((ext_vector_type(8)))  float    v8f;

// ---------------- monotone float<->uint key (order-preserving) ----------------
__device__ __forceinline__ unsigned f2key(float v) {
  unsigned b = __float_as_uint(v);
  return (b & 0x80000000u) ? ~b : (b | 0x80000000u);
}

// ---------------- weight prep: f32 [K,N] -> f16 transposed [N,K] --------------
__global__ void k_w16t(const float* __restrict__ src, _Float16* __restrict__ dst,
                       int K, int N) {
  int t = blockIdx.x * 256 + threadIdx.x;
  if (t >= K * N) return;
  int k = t / N, n = t % N;
  dst[(size_t)n * K + k] = (_Float16)src[t];
}

// ---------------- embedding: relu(x[N,6] @ W[6,64] + b) ----------------------
__global__ void k_embed(const float* __restrict__ x, const float* __restrict__ W,
                        const float* __restrict__ b, float* __restrict__ h, int N) {
  int t = blockIdx.x * 256 + threadIdx.x;
  if (t >= N * 64) return;
  int n = t >> 6, c = t & 63;
  float acc = b[c];
#pragma unroll
  for (int k = 0; k < 6; ++k) acc += x[n * 6 + k] * W[k * 64 + c];
  h[t] = fmaxf(acc, 0.f);
}

__global__ void k_fill_u32(unsigned* p, unsigned v, int n) {
  int i = blockIdx.x * 256 + threadIdx.x;
  if (i < n) p[i] = v;
}

// ---------------- GAPL edge kernel: WMMA feat GEMM + VALU geom + atomic segmax
// wave handles 16 edges; A = f16(h[src]-h[dst]) in 16x16x32 A layout;
// B = transposed f16 weights (contig 16 halfs per lane).
template <int H>
__global__ __launch_bounds__(256) void k_gapl_edges(
    const float* __restrict__ pos, const float* __restrict__ h,
    const int* __restrict__ esrc, const int* __restrict__ edst,
    const float* __restrict__ Wg, const float* __restrict__ bg,
    const _Float16* __restrict__ Wft, const float* __restrict__ bfv,
    unsigned* __restrict__ agg) {
  constexpr int KC = H / 32, NT = H / 16;
  __shared__ int   sS[8][16];
  __shared__ int   sD[8][16];
  __shared__ float sP[8][48];
  const int wave = threadIdx.x >> 5, lane = threadIdx.x & 31;
  const int half = lane >> 4, ln = lane & 15;
  const int e0 = (blockIdx.x * 8 + wave) * 16;

  __builtin_prefetch(Wft, 0, 3);  // global_prefetch_b8 of hot weight tile

  if (lane < 16) {
    int s = esrc[e0 + lane], d = edst[e0 + lane];
    sS[wave][lane] = s;
    sD[wave][lane] = d;
    sP[wave][lane * 3 + 0] = pos[s * 3 + 0] - pos[d * 3 + 0];
    sP[wave][lane * 3 + 1] = pos[s * 3 + 1] - pos[d * 3 + 1];
    sP[wave][lane * 3 + 2] = pos[s * 3 + 2] - pos[d * 3 + 2];
  }
  __syncthreads();

  const float* hs = h + (size_t)sS[wave][ln] * H;
  const float* hd = h + (size_t)sD[wave][ln] * H;
  v16h A[KC];
#pragma unroll
  for (int kc = 0; kc < KC; ++kc) {
    const int ks = kc * 32 + half * 16;
#pragma unroll
    for (int j = 0; j < 8; ++j) {
      A[kc][j]     = (_Float16)(hs[ks + j]      - hd[ks + j]);
      A[kc][j + 8] = (_Float16)(hs[ks + 8 + j] - hd[ks + 8 + j]);
    }
  }

#pragma unroll
  for (int nt = 0; nt < NT; ++nt) {
    const int n = nt * 16 + ln;
    v8f c = {};
#pragma unroll
    for (int kc = 0; kc < KC; ++kc) {
      const v16h b = *(const v16h*)(Wft + (size_t)n * H + kc * 32 + half * 16);
      c = __builtin_amdgcn_wmma_f32_16x16x32_f16(false, A[kc], false, b,
                                                 (short)0, c, false, false);
    }
    const float bfn = bfv[n], bgn = bg[n];
    const float w0 = Wg[n], w1 = Wg[H + n], w2 = Wg[2 * H + n];
#pragma unroll
    for (int r = 0; r < 8; ++r) {
      const int m = r + 8 * half;
      float feat = fmaxf(c[r] + bfn, 0.f);
      float g = fmaxf(sP[wave][m * 3] * w0 + sP[wave][m * 3 + 1] * w1 +
                          sP[wave][m * 3 + 2] * w2 + bgn, 0.f);
      float v = 0.5f * feat + 0.5f * g + h[(size_t)sS[wave][m] * H + n];
      atomicMax(agg + (size_t)sD[wave][m] * H + n, f2key(v));
    }
  }
}

// ---------------- node: h_out = h_in + decode(segmax key) --------------------
__global__ void k_gapl_node(const float* __restrict__ hin,
                            const unsigned* __restrict__ agg,
                            float* __restrict__ hout, int n) {
  int i = blockIdx.x * 256 + threadIdx.x;
  if (i >= n) return;
  unsigned k = agg[i];
  float a = 0.f;
  if (k) {
    unsigned b = (k & 0x80000000u) ? (k & 0x7FFFFFFFu) : ~k;
    a = __uint_as_float(b);
  }
  hout[i] = hin[i] + a;
}

// ---------------- fused ResMLP: h += relu(h@W1+b1)@W2+b2 ---------------------
template <int H, int WAVES>
__global__ __launch_bounds__(WAVES * 32) void k_resblk(
    float* __restrict__ hio, const _Float16* __restrict__ W1t,
    const float* __restrict__ b1, const _Float16* __restrict__ W2t,
    const float* __restrict__ b2) {
  constexpr int KC = H / 32, NT = H / 16;
  __shared__ _Float16 sT[WAVES][16 * H];
  const int wave = threadIdx.x >> 5, lane = threadIdx.x & 31;
  const int half = lane >> 4, ln = lane & 15;
  const size_t row0 = (size_t)(blockIdx.x * WAVES + wave) * 16;
  const float* arow = hio + (row0 + ln) * H;

  v16h A[KC];
#pragma unroll
  for (int kc = 0; kc < KC; ++kc) {
    const int ks = kc * 32 + half * 16;
#pragma unroll
    for (int j = 0; j < 8; ++j) {
      A[kc][j]     = (_Float16)arow[ks + j];
      A[kc][j + 8] = (_Float16)arow[ks + 8 + j];
    }
  }
#pragma unroll
  for (int nt = 0; nt < NT; ++nt) {
    const int n = nt * 16 + ln;
    v8f c = {};
#pragma unroll
    for (int kc = 0; kc < KC; ++kc) {
      const v16h b = *(const v16h*)(W1t + (size_t)n * H + kc * 32 + half * 16);
      c = __builtin_amdgcn_wmma_f32_16x16x32_f16(false, A[kc], false, b,
                                                 (short)0, c, false, false);
    }
    const float bn = b1[n];
#pragma unroll
    for (int r = 0; r < 8; ++r)
      sT[wave][(r + 8 * half) * H + n] = (_Float16)fmaxf(c[r] + bn, 0.f);
  }
  __syncthreads();

  v16h A2[KC];
#pragma unroll
  for (int kc = 0; kc < KC; ++kc) {
    const int ks = kc * 32 + half * 16;
#pragma unroll
    for (int j = 0; j < 8; ++j) {
      A2[kc][j]     = sT[wave][ln * H + ks + j];
      A2[kc][j + 8] = sT[wave][ln * H + ks + 8 + j];
    }
  }
#pragma unroll
  for (int nt = 0; nt < NT; ++nt) {
    const int n = nt * 16 + ln;
    v8f c = {};
#pragma unroll
    for (int kc = 0; kc < KC; ++kc) {
      const v16h b = *(const v16h*)(W2t + (size_t)n * H + kc * 32 + half * 16);
      c = __builtin_amdgcn_wmma_f32_16x16x32_f16(false, A2[kc], false, b,
                                                 (short)0, c, false, false);
    }
    const float bn = b2[n];
#pragma unroll
    for (int r = 0; r < 8; ++r) {
      const size_t o = (row0 + r + 8 * half) * H + n;
      hio[o] = hio[o] + c[r] + bn;
    }
  }
}

// ---------------- MLP: out = relu(in[rows,HK] @ W + b), HN wide --------------
template <int HK, int HN>
__global__ __launch_bounds__(256) void k_mlp(const float* __restrict__ in,
                                             float* __restrict__ out,
                                             const _Float16* __restrict__ Wt,
                                             const float* __restrict__ bias) {
  constexpr int KC = HK / 32, NT = HN / 16;
  const int wave = threadIdx.x >> 5, lane = threadIdx.x & 31;
  const int half = lane >> 4, ln = lane & 15;
  const size_t row0 = (size_t)(blockIdx.x * 8 + wave) * 16;
  const float* arow = in + (row0 + ln) * HK;

  v16h A[KC];
#pragma unroll
  for (int kc = 0; kc < KC; ++kc) {
    const int ks = kc * 32 + half * 16;
#pragma unroll
    for (int j = 0; j < 8; ++j) {
      A[kc][j]     = (_Float16)arow[ks + j];
      A[kc][j + 8] = (_Float16)arow[ks + 8 + j];
    }
  }
#pragma unroll
  for (int nt = 0; nt < NT; ++nt) {
    const int n = nt * 16 + ln;
    v8f c = {};
#pragma unroll
    for (int kc = 0; kc < KC; ++kc) {
      const v16h b = *(const v16h*)(Wt + (size_t)n * HK + kc * 32 + half * 16);
      c = __builtin_amdgcn_wmma_f32_16x16x32_f16(false, A[kc], false, b,
                                                 (short)0, c, false, false);
    }
    const float bn = bias[n];
#pragma unroll
    for (int r = 0; r < 8; ++r)
      out[(row0 + r + 8 * half) * HN + n] = fmaxf(c[r] + bn, 0.f);
  }
}

// ---------------- pooling gathers --------------------------------------------
__global__ void k_gather_feat(const float* __restrict__ h, const int* __restrict__ perm,
                              float* __restrict__ out, int np, int H) {
  int t = blockIdx.x * 256 + threadIdx.x;
  if (t >= np * H) return;
  int i = t / H, c = t % H;
  out[t] = h[(size_t)perm[i] * H + c];
}
__global__ void k_gather_pos(const float* __restrict__ pos, const int* __restrict__ perm,
                             float* __restrict__ out, int np) {
  int t = blockIdx.x * 256 + threadIdx.x;
  if (t >= np) return;
  int p = perm[t];
  out[t * 3 + 0] = pos[p * 3 + 0];
  out[t * 3 + 1] = pos[p * 3 + 1];
  out[t * 3 + 2] = pos[p * 3 + 2];
}

// ---------------- edge similarity pruning ------------------------------------
__global__ void k_norms(const float* __restrict__ xp, float* __restrict__ inv,
                        int np, int H) {
  int i = blockIdx.x * 256 + threadIdx.x;
  if (i >= np) return;
  const float* p = xp + (size_t)i * H;
  float s = 1e-8f;
  for (int c = 0; c < H; ++c) s += p[c] * p[c];
  inv[i] = rsqrtf(s);
}
__global__ void k_sim(const float* __restrict__ xp, const float* __restrict__ inv,
                      const int* __restrict__ ea, const int* __restrict__ eb,
                      int E, int H, unsigned* __restrict__ keys) {
  int e = blockIdx.x * 256 + threadIdx.x;
  if (e >= E) return;
  int a = ea[e], b = eb[e];
  const float* pa = xp + (size_t)a * H;
  const float* pb = xp + (size_t)b * H;
  float s = 0.f;
  for (int c = 0; c < H; ++c) s += pa[c] * pb[c];
  s *= inv[a] * inv[b];
  keys[e] = f2key(s);
}
__global__ void k_state_init(unsigned* state, unsigned K) {
  if (threadIdx.x == 0 && blockIdx.x == 0) { state[0] = 0u; state[1] = K; state[2] = 0u; }
}
__global__ void k_hist(const unsigned* __restrict__ keys, int E, int shift, int width,
                       const unsigned* __restrict__ state, unsigned* __restrict__ hist) {
  int e = blockIdx.x * 256 + threadIdx.x;
  if (e >= E) return;
  unsigned k = keys[e];
  int ps = shift + width;
  if (ps >= 32 || (k >> ps) == state[0])
    atomicAdd(hist + ((k >> shift) & ((1u << width) - 1u)), 1u);
}
__global__ void k_pick(const unsigned* __restrict__ hist, int bins, int width,
                       unsigned* __restrict__ state) {
  if (threadIdx.x || blockIdx.x) return;
  unsigned need = state[1], cum = 0;
  for (int b = bins - 1; b >= 0; --b) {
    unsigned c = hist[b];
    if (cum + c >= need) {
      state[0] = (state[0] << width) | (unsigned)b;
      state[1] = need - cum;
      break;
    }
    cum += c;
  }
}
__global__ __launch_bounds__(256) void k_count(const unsigned* __restrict__ keys,
                                               const unsigned* __restrict__ state,
                                               unsigned* __restrict__ blkc) {
  __shared__ unsigned sg, sq;
  if (threadIdx.x == 0) { sg = 0; sq = 0; }
  __syncthreads();
  unsigned T = state[0];
  int base = blockIdx.x * 4096 + threadIdx.x * 16;
  unsigned g = 0, q = 0;
  for (int j = 0; j < 16; ++j) {
    unsigned k = keys[base + j];
    g += (k > T); q += (k == T);
  }
  atomicAdd(&sg, g); atomicAdd(&sq, q);
  __syncthreads();
  if (threadIdx.x == 0) { blkc[2 * blockIdx.x] = sg; blkc[2 * blockIdx.x + 1] = sq; }
}
__global__ void k_scan(const unsigned* __restrict__ blkc, int nb,
                       unsigned* __restrict__ blko, unsigned* __restrict__ state) {
  if (threadIdx.x || blockIdx.x) return;
  unsigned rg = 0, rq = 0;
  for (int b = 0; b < nb; ++b) {
    blko[2 * b] = rg; blko[2 * b + 1] = rq;
    rg += blkc[2 * b]; rq += blkc[2 * b + 1];
  }
  state[2] = rg;
}
__global__ __launch_bounds__(256) void k_scatter(
    const unsigned* __restrict__ keys, const int* __restrict__ ea,
    const int* __restrict__ eb, int K, const unsigned* __restrict__ state,
    const unsigned* __restrict__ blko, int* __restrict__ osrc, int* __restrict__ odst) {
  __shared__ unsigned sgx[256], sqx[256];
  const unsigned T = state[0], totg = state[2];
  const int t = threadIdx.x;
  const int base = blockIdx.x * 4096 + t * 16;
  unsigned g = 0, q = 0;
  for (int j = 0; j < 16; ++j) {
    unsigned k = keys[base + j];
    g += (k > T); q += (k == T);
  }
  sgx[t] = g; sqx[t] = q;
  __syncthreads();
  for (int d = 1; d < 256; d <<= 1) {
    unsigned vg = 0, vq = 0;
    if (t >= d) { vg = sgx[t - d]; vq = sqx[t - d]; }
    __syncthreads();
    sgx[t] += vg; sqx[t] += vq;
    __syncthreads();
  }
  unsigned og = blko[2 * blockIdx.x] + sgx[t] - g;
  unsigned oq = totg + blko[2 * blockIdx.x + 1] + sqx[t] - q;
  for (int j = 0; j < 16; ++j) {
    int e = base + j;
    unsigned k = keys[e];
    if (k > T) { osrc[og] = ea[e]; odst[og] = eb[e]; ++og; }
    else if (k == T) {
      if (oq < (unsigned)K) { osrc[oq] = ea[e]; odst[oq] = eb[e]; }
      ++oq;
    }
  }
}
__global__ void k_i2f(const int* __restrict__ in, float* __restrict__ out, int n) {
  int i = blockIdx.x * 256 + threadIdx.x;
  if (i < n) out[i] = (float)in[i];
}

// =============================================================================
extern "C" void kernel_launch(void* const* d_in, const int* in_sizes, int n_in,
                              void* d_out, int out_size, void* d_ws, size_t ws_size,
                              hipStream_t stream) {
  (void)in_sizes; (void)n_in; (void)out_size;
  const int N0 = 65536, E0 = 1048576;
  const int N1 = 32768, E1in = 524288, K1 = 262144;
  const int N2 = 16384, E2in = 262144, K2 = 131072;

  if (ws_size < (size_t)56 << 20) return;  // need ~55 MB scratch

  auto F = [&](int i) { return (const float*)d_in[i]; };
  auto I = [&](int i) { return (const int*)d_in[i]; };
  const float* pos = F(0);
  const int* ei = I(2);
  const int* perm1 = I(4); const int* e1s = I(5); const int* e1d = I(5) + E1in;
  const int* perm2 = I(6); const int* e2s = I(7); const int* e2d = I(7) + E2in;
  float* out = (float*)d_out;

  // output section offsets (floats), return order: pos tuple, x tuple, eidx tuple
  const size_t o_pos0 = 0, o_pos1 = 196608, o_pos2 = 294912;
  const size_t o_x0 = 344064, o_x1 = 4538368, o_x2 = 8732672;
  const size_t o_e0 = 12926976, o_e1 = 15024128, o_e2 = 15548416;

  char* W = (char*)d_ws;
  float*    RA   = (float*)(W);
  float*    RB   = (float*)(W + ((size_t)16 << 20));
  unsigned* AGG  = (unsigned*)(W + ((size_t)32 << 20));
  _Float16* WT   = (_Float16*)(W + ((size_t)48 << 20));
  unsigned* SIMK = (unsigned*)(W + ((size_t)49 << 20));
  int*      PR1  = (int*)(W + ((size_t)51 << 20));
  int*      PR2  = (int*)(W + ((size_t)53 << 20));
  float*    INV  = (float*)(W + ((size_t)54 << 20));
  unsigned* HIST = (unsigned*)(W + ((size_t)54 << 20) + (1 << 18));
  unsigned* STATE = HIST + 2048;
  unsigned* BLKC = STATE + 8;
  unsigned* BLKO = BLKC + 512;

  // ---- weight prep: f16 transposed copies of all GEMM weights ----
  static const int widx[29] = {12, 24, 36, 14, 16, 18, 20, 26, 28, 30, 32,
                               38, 40, 42, 44, 46, 48, 50, 52, 54, 56, 58,
                               60, 62, 64, 66, 68, 70, 72};
  static const int wK[29] = {64, 64, 128, 64, 64, 64, 64, 64, 64, 64, 64,
                             128, 128, 128, 128, 64, 64, 64, 64, 64, 128, 128,
                             128, 128, 128, 256, 256, 256, 256};
  static const int wN[29] = {64, 64, 128, 64, 64, 64, 64, 64, 64, 64, 64,
                             128, 128, 128, 128, 64, 64, 64, 64, 128, 128, 128,
                             128, 128, 256, 256, 256, 256, 256};
  size_t woff[29], acc = 0;
  for (int i = 0; i < 29; ++i) { woff[i] = acc; acc += (size_t)wK[i] * wN[i]; }
  for (int i = 0; i < 29; ++i) {
    int KN = wK[i] * wN[i];
    k_w16t<<<(KN + 255) / 256, 256, 0, stream>>>(F(widx[i]), WT + woff[i], wK[i], wN[i]);
  }

  auto radix_select = [&](const unsigned* keys, int E, unsigned K, const int* ea,
                          const int* eb, int* osrc, int* odst) {
    k_state_init<<<1, 1, 0, stream>>>(STATE, K);
    const int shifts[3] = {21, 10, 0}, widths[3] = {11, 11, 10};
    for (int p = 0; p < 3; ++p) {
      k_fill_u32<<<8, 256, 0, stream>>>(HIST, 0u, 2048);
      k_hist<<<E / 256, 256, 0, stream>>>(keys, E, shifts[p], widths[p], STATE, HIST);
      k_pick<<<1, 1, 0, stream>>>(HIST, 1 << widths[p], widths[p], STATE);
    }
    int nb = E / 4096;
    k_count<<<nb, 256, 0, stream>>>(keys, STATE, BLKC);
    k_scan<<<1, 1, 0, stream>>>(BLKC, nb, BLKO, STATE);
    k_scatter<<<nb, 256, 0, stream>>>(keys, ea, eb, (int)K, STATE, BLKO, osrc, odst);
  };

  // ================= Level 0 =================
  k_embed<<<(N0 * 64) / 256, 256, 0, stream>>>(F(1), F(8), F(9), RA, N0);
  k_fill_u32<<<(N0 * 64) / 256, 256, 0, stream>>>(AGG, 0u, N0 * 64);
  k_gapl_edges<64><<<E0 / 128, 256, 0, stream>>>(pos, RA, ei, ei + E0, F(10), F(11),
                                                 WT + woff[0], F(13), AGG);
  k_gapl_node<<<(N0 * 64) / 256, 256, 0, stream>>>(RA, AGG, RB, N0 * 64);
  k_resblk<64, 8><<<N0 / 128, 256, 0, stream>>>(RB, WT + woff[3], F(15), WT + woff[4], F(17));
  k_resblk<64, 8><<<N0 / 128, 256, 0, stream>>>(RB, WT + woff[5], F(19), WT + woff[6], F(21));
  k_resblk<64, 8><<<N0 / 128, 256, 0, stream>>>(RB, WT + woff[15], F(47), WT + woff[16], F(49));
  k_resblk<64, 8><<<N0 / 128, 256, 0, stream>>>(RB, WT + woff[17], F(51), WT + woff[18], F(53));
  hipMemcpyAsync(out + o_x0, RB, (size_t)N0 * 64 * 4, hipMemcpyDeviceToDevice, stream);

  // ================= Level 1 =================
  k_gather_feat<<<(N1 * 64) / 256, 256, 0, stream>>>(RB, perm1, RA, N1, 64);
  k_gather_pos<<<(N1 + 255) / 256, 256, 0, stream>>>(pos, perm1, out + o_pos1, N1);
  k_norms<<<(N1 + 255) / 256, 256, 0, stream>>>(RA, INV, N1, 64);
  k_sim<<<E1in / 256, 256, 0, stream>>>(RA, INV, e1s, e1d, E1in, 64, SIMK);
  radix_select(SIMK, E1in, (unsigned)K1, e1s, e1d, PR1, PR1 + K1);
  k_fill_u32<<<(N1 * 64) / 256, 256, 0, stream>>>(AGG, 0u, N1 * 64);
  k_gapl_edges<64><<<K1 / 128, 256, 0, stream>>>(out + o_pos1, RA, PR1, PR1 + K1,
                                                 F(22), F(23), WT + woff[1], F(25), AGG);
  k_gapl_node<<<(N1 * 64) / 256, 256, 0, stream>>>(RA, AGG, RB, N1 * 64);
  k_resblk<64, 8><<<N1 / 128, 256, 0, stream>>>(RB, WT + woff[7], F(27), WT + woff[8], F(29));
  k_resblk<64, 8><<<N1 / 128, 256, 0, stream>>>(RB, WT + woff[9], F(31), WT + woff[10], F(33));
  k_mlp<64, 128><<<N1 / 128, 256, 0, stream>>>(RB, RA, WT + woff[19], F(55));
  k_resblk<128, 8><<<N1 / 128, 256, 0, stream>>>(RA, WT + woff[20], F(57), WT + woff[21], F(59));
  k_resblk<128, 8><<<N1 / 128, 256, 0, stream>>>(RA, WT + woff[22], F(61), WT + woff[23], F(63));
  hipMemcpyAsync(out + o_x1, RA, (size_t)N1 * 128 * 4, hipMemcpyDeviceToDevice, stream);
  k_i2f<<<(2 * K1) / 256, 256, 0, stream>>>(PR1, out + o_e1, 2 * K1);

  // ================= Level 2 =================
  k_gather_feat<<<(N2 * 128) / 256, 256, 0, stream>>>(RA, perm2, RB, N2, 128);
  k_gather_pos<<<(N2 + 255) / 256, 256, 0, stream>>>(out + o_pos1, perm2, out + o_pos2, N2);
  k_norms<<<(N2 + 255) / 256, 256, 0, stream>>>(RB, INV, N2, 128);
  k_sim<<<E2in / 256, 256, 0, stream>>>(RB, INV, e2s, e2d, E2in, 128, SIMK);
  radix_select(SIMK, E2in, (unsigned)K2, e2s, e2d, PR2, PR2 + K2);
  k_fill_u32<<<(N2 * 128) / 256, 256, 0, stream>>>(AGG, 0u, N2 * 128);
  k_gapl_edges<128><<<K2 / 128, 256, 0, stream>>>(out + o_pos2, RB, PR2, PR2 + K2,
                                                  F(34), F(35), WT + woff[2], F(37), AGG);
  k_gapl_node<<<(N2 * 128) / 256, 256, 0, stream>>>(RB, AGG, RA, N2 * 128);
  k_resblk<128, 8><<<N2 / 128, 256, 0, stream>>>(RA, WT + woff[11], F(39), WT + woff[12], F(41));
  k_resblk<128, 8><<<N2 / 128, 256, 0, stream>>>(RA, WT + woff[13], F(43), WT + woff[14], F(45));
  k_mlp<128, 256><<<N2 / 128, 256, 0, stream>>>(RA, RB, WT + woff[24], F(65));
  k_resblk<256, 4><<<N2 / 64, 128, 0, stream>>>(RB, WT + woff[25], F(67), WT + woff[26], F(69));
  k_resblk<256, 4><<<N2 / 64, 128, 0, stream>>>(RB, WT + woff[27], F(71), WT + woff[28], F(73));
  hipMemcpyAsync(out + o_x2, RB, (size_t)N2 * 256 * 4, hipMemcpyDeviceToDevice, stream);
  k_i2f<<<(2 * K2) / 256, 256, 0, stream>>>(PR2, out + o_e2, 2 * K2);

  // ================= passthrough outputs =================
  hipMemcpyAsync(out + o_pos0, pos, (size_t)N0 * 3 * 4, hipMemcpyDeviceToDevice, stream);
  k_i2f<<<(2 * E0) / 256, 256, 0, stream>>>(ei, out + o_e0, 2 * E0);
}
